// MSDeformAttn_73761768341600
// MI455X (gfx1250) — matrix operand order repaired and checked by reading, not compiled, not run on hardware
//
#include <hip/hip_runtime.h>
#include <hip/hip_bf16.h>
#include <math.h>

// ---------------------------------------------------------------------------
// MS Deformable Attention on gfx1250 (MI455X).
// Static problem constants (spatial shapes / level starts are static numpy in
// the reference, so they are baked in).
// ---------------------------------------------------------------------------
#define NB    2
#define LQ    13294
#define LIN   13294
#define CDIM  256
#define KDIM  256          // K is 256 for every GEMM in this model
#define HN    8
#define DDIM  32
#define LN    4
#define PN    4
#define MROWS (NB * LQ)    // 26588 rows for every GEMM

typedef __attribute__((ext_vector_type(16))) _Float16 v16h;
typedef __attribute__((ext_vector_type(8)))  _Float16 v8h;
typedef __attribute__((ext_vector_type(4)))  _Float16 v4h;
typedef __attribute__((ext_vector_type(8)))  float    v8f;

union CFrag { v8f  v; float e[8]; };
union BFrag { v16h v; v8h  h[2]; };

// ---------------------------------------------------------------------------
// WMMA GEMM: C[M,Nw] = A[M,256] * B[256,Nw] + bias[Nw]  (fp32 memory, f16
// tensor path, f32 accumulate). Block = 256 thr = 8 waves; block tile 64x32;
// each wave owns one 16x16 tile; K stepped by 32, fully unrolled (8 WMMAs).
//
// LDS formats are chosen so fragments load as wide DS ops:
//  sA: fragment-major [rowSubtile][lane][16 halves] -> A frag = 1x 32B read.
//  sBt: column-major [n][k] halves                  -> B frag = 2x 16B reads.
// ---------------------------------------------------------------------------
__global__ __launch_bounds__(256)
void gemm_wmma_f16(const float* __restrict__ A, const float* __restrict__ B,
                   const float* __restrict__ bias, float* __restrict__ C,
                   int M, int Nw) {
  __shared__ __align__(32) _Float16 sA[4][32][16];  // 4KB, fragment-major
  __shared__ __align__(16) _Float16 sBt[32][40];    // 2.5KB, [n][k] padded

  const int t    = threadIdx.x;
  const int wave = t >> 5;
  const int lane = t & 31;
  const int rowSub  = wave & 3;          // which 16-row subtile (0..3)
  const int colTile = (wave >> 2) * 16;  // 0 or 16
  const int blockRow = blockIdx.x * 64;
  const int blockCol = blockIdx.y * 32;

  // ---- A staging map: thread t loads row (t>>2), k-chunk (t&3)*8 ----
  const int a_row   = t >> 2;            // 0..63 global-tile row
  const int a_chunk = t & 3;             // 0..3 (8 k each)
  const int a_sub   = a_row >> 4;        // row subtile
  const int a_m     = a_row & 15;        // row within subtile
  // WMMA A layout: chunk0(K0-7)->lane m half0-7; chunk1(K8-15)->lane m+16 h0-7
  //                chunk2(K16-23)->lane m h8-15; chunk3(K24-31)->lane m+16 h8-15
  const int a_lane  = (a_chunk & 1) ? (a_m + 16) : a_m;
  const int a_half  = (a_chunk >> 1) * 8;
  const int gr      = blockRow + a_row;

  // ---- B staging map: 64 threads each own a 4x4 f32 sub-block ----
  const bool b_act = (t < 64);
  const int  b_kq  = (t >> 3) * 4;       // k quad base (0..28)
  const int  b_nq  = (t & 7) * 4;        // n quad base (0..28)

  CFrag acc;
#pragma unroll
  for (int r = 0; r < 8; ++r) acc.e[r] = 0.0f;

  const int mrow = lane & 15;
  const int ncol = colTile + (lane & 15);
  const int kb   = (lane < 16) ? 0 : 8;

#pragma unroll
  for (int k0 = 0; k0 < KDIM; k0 += 32) {
    // ---- stage A: one b128 global load pair -> one 16B LDS store ----
    if (gr < M) {
      const float4* p = (const float4*)(A + (size_t)gr * KDIM + k0 + a_chunk * 8);
      float4 f0 = p[0];
      float4 f1 = p[1];
      v8h pk;
      pk[0] = (_Float16)f0.x; pk[1] = (_Float16)f0.y;
      pk[2] = (_Float16)f0.z; pk[3] = (_Float16)f0.w;
      pk[4] = (_Float16)f1.x; pk[5] = (_Float16)f1.y;
      pk[6] = (_Float16)f1.z; pk[7] = (_Float16)f1.w;
      *(v8h*)&sA[a_sub][a_lane][a_half] = pk;
      if (k0 + 32 < KDIM)   // -> global_prefetch_b8
        __builtin_prefetch(A + (size_t)gr * KDIM + k0 + 32 + a_chunk * 8, 0, 1);
    } else {
      *(v8h*)&sA[a_sub][a_lane][a_half] = (v8h)(_Float16)0.0f;
    }
    // ---- stage B: 4x4 f32 sub-block, register transpose, 4x 8B LDS stores --
    if (b_act) {
      float r0[4], r1[4], r2[4], r3[4];
      {
        const float* pb = B + (size_t)(k0 + b_kq) * Nw + blockCol + b_nq;
        float4 q0 = *(const float4*)(pb + 0 * Nw);
        float4 q1 = *(const float4*)(pb + 1 * Nw);
        float4 q2 = *(const float4*)(pb + 2 * Nw);
        float4 q3 = *(const float4*)(pb + 3 * Nw);
        r0[0]=q0.x; r0[1]=q0.y; r0[2]=q0.z; r0[3]=q0.w;
        r1[0]=q1.x; r1[1]=q1.y; r1[2]=q1.z; r1[3]=q1.w;
        r2[0]=q2.x; r2[1]=q2.y; r2[2]=q2.z; r2[3]=q2.w;
        r3[0]=q3.x; r3[1]=q3.y; r3[2]=q3.z; r3[3]=q3.w;
        if (k0 + 32 < KDIM)
          __builtin_prefetch(pb + (size_t)32 * Nw, 0, 1);
      }
#pragma unroll
      for (int j = 0; j < 4; ++j) {
        v4h pk;
        pk[0] = (_Float16)r0[j]; pk[1] = (_Float16)r1[j];
        pk[2] = (_Float16)r2[j]; pk[3] = (_Float16)r3[j];
        *(v4h*)&sBt[b_nq + j][b_kq] = pk;   // [n][k], 8B aligned
      }
    }
    __syncthreads();

    // ---- fragments: wide DS loads only ----
    v16h af = *(const v16h*)&sA[rowSub][lane][0];      // 32B -> 2x ds_load_b128
    BFrag bf;
    bf.h[0] = *(const v8h*)&sBt[ncol][kb + 0];         // 16B ds_load_b128
    bf.h[1] = *(const v8h*)&sBt[ncol][kb + 16];        // 16B ds_load_b128

    acc.v = __builtin_amdgcn_wmma_f32_16x16x32_f16(
        /*neg_a=*/false, af, /*neg_b=*/false, bf.v,
        /*c_mod=*/(short)0, acc.v, /*reuse_a=*/false, /*reuse_b=*/false);
    __syncthreads();
  }

  // C/D 16x16 f32 layout: VGPR r -> (M=r, N=lane) lanes 0-15;
  //                       (M=8+r, N=lane-16) lanes 16-31.
  const int moff = (lane < 16) ? 0 : 8;
  const int nn = blockCol + colTile + (lane & 15);
  const float bb = bias ? bias[nn] : 0.0f;
#pragma unroll
  for (int r = 0; r < 8; ++r) {
    const int m = blockRow + rowSub * 16 + moff + r;
    if (m < M) C[(size_t)m * Nw + nn] = acc.e[r] + bb;
  }
}

// ---------------------------------------------------------------------------
// Softmax over contiguous rows of 16 (per (n,q,h): the L*P logits). In-place.
// ---------------------------------------------------------------------------
__global__ __launch_bounds__(256)
void softmax16(float* __restrict__ logits, int rows) {
  const int t = blockIdx.x * 256 + threadIdx.x;
  if (t >= rows) return;
  float* p = logits + (size_t)t * 16;
  float v[16];
  float mx = -3.402823e38f;
#pragma unroll
  for (int i = 0; i < 16; ++i) { v[i] = p[i]; mx = fmaxf(mx, v[i]); }
  float s = 0.0f;
#pragma unroll
  for (int i = 0; i < 16; ++i) { v[i] = __expf(v[i] - mx); s += v[i]; }
  const float inv = 1.0f / s;
#pragma unroll
  for (int i = 0; i < 16; ++i) p[i] = v[i] * inv;
}

// ---------------------------------------------------------------------------
// Deformable bilinear sampling. One 32-lane group per (n,q,h); lane = channel
// d, so every corner gather of value[n, idx, h, :] is one contiguous 128B
// line (coalesced; `value` is L2-resident: 26 MB vs the 192 MB global L2).
// Block = 256 threads = 8 (n,q,h) units; grid.x = MROWS.
// ---------------------------------------------------------------------------
__global__ __launch_bounds__(256)
void deform_sample(const float* __restrict__ refp, const float* __restrict__ off,
                   const float* __restrict__ attn, const float* __restrict__ value,
                   float* __restrict__ samp) {
  const int lvlH[LN] = {100, 50, 25, 13};
  const int lvlW[LN] = {100, 50, 25, 13};
  const int lvlS[LN] = {0, 10000, 12500, 13125};

  const int unit = blockIdx.x * 8 + (threadIdx.x >> 5);  // (n*LQ+q)*HN + h
  const int d  = threadIdx.x & 31;
  const int h  = unit & (HN - 1);
  const int nq = unit >> 3;                              // n*LQ + q
  const int n  = nq / LQ;

  const float* vb = value + (size_t)n * LIN * CDIM + h * DDIM + d;
  float acc = 0.0f;

#pragma unroll
  for (int l = 0; l < LN; ++l) {
    const int Hi = lvlH[l], Wi = lvlW[l], s = lvlS[l];
    const float Wf = (float)Wi, Hf = (float)Hi;
    const float rx = refp[((size_t)nq * LN + l) * 2 + 0];
    const float ry = refp[((size_t)nq * LN + l) * 2 + 1];
#pragma unroll
    for (int p = 0; p < PN; ++p) {
      const size_t ob = (size_t)nq * (HN * LN * PN * 2) +
                        (size_t)(((h * LN + l) * PN + p) * 2);
      const float ox = off[ob + 0];
      const float oy = off[ob + 1];
      const float aw = attn[(size_t)nq * (HN * LN * PN) + h * (LN * PN) + l * PN + p];
      // loc = ref + off/(W,H); pixel = loc*(W,H) - 0.5  =>  ref*W + off - 0.5
      const float x = rx * Wf + ox - 0.5f;
      const float y = ry * Hf + oy - 0.5f;
      const float x0f = floorf(x), y0f = floorf(y);
      const float lx = x - x0f, ly = y - y0f;
      const int x0 = (int)x0f, y0 = (int)y0f;
      const float w00 = (1.0f - ly) * (1.0f - lx);
      const float w01 = (1.0f - ly) * lx;
      const float w10 = ly * (1.0f - lx);
      const float w11 = ly * lx;
      const float cw[4]  = {w00, w01, w10, w11};
      const int   cdx[4] = {0, 1, 0, 1};
      const int   cdy[4] = {0, 0, 1, 1};
#pragma unroll
      for (int c = 0; c < 4; ++c) {
        const int xi = x0 + cdx[c];
        const int yi = y0 + cdy[c];
        if (xi >= 0 && xi < Wi && yi >= 0 && yi < Hi) {
          acc += aw * cw[c] * vb[(size_t)(s + yi * Wi + xi) * CDIM];
        }
      }
    }
  }
  samp[(size_t)nq * CDIM + h * DDIM + d] = acc;
}

// ---------------------------------------------------------------------------
// Launch: value GEMM, offset GEMM, attn GEMM, softmax, sampling, output GEMM.
// d_in order: query, reference_points, input_flatten, spatial, level_start,
//             W_off, b_off, W_attn, b_attn, Wv, bv, Wo, bo
// ---------------------------------------------------------------------------
extern "C" void kernel_launch(void* const* d_in, const int* in_sizes, int n_in,
                              void* d_out, int out_size, void* d_ws, size_t ws_size,
                              hipStream_t stream) {
  (void)in_sizes; (void)n_in; (void)out_size; (void)ws_size;
  const float* query  = (const float*)d_in[0];
  const float* refpts = (const float*)d_in[1];
  const float* input  = (const float*)d_in[2];
  const float* W_off  = (const float*)d_in[5];
  const float* b_off  = (const float*)d_in[6];
  const float* W_attn = (const float*)d_in[7];
  const float* b_attn = (const float*)d_in[8];
  const float* Wv     = (const float*)d_in[9];
  const float* bv     = (const float*)d_in[10];
  const float* Wo     = (const float*)d_in[11];
  const float* bo     = (const float*)d_in[12];
  float* out = (float*)d_out;

  // Workspace layout (floats): value | off | attn | samp  (~95 MB total)
  float* value = (float*)d_ws;
  float* off   = value + (size_t)MROWS * CDIM;
  float* attn  = off   + (size_t)MROWS * CDIM;
  float* samp  = attn  + (size_t)MROWS * (HN * LN * PN);

  const dim3 blk(256);
  const dim3 g256((MROWS + 63) / 64, CDIM / 32);
  const dim3 g128((MROWS + 63) / 64, (HN * LN * PN) / 32);

  // 1) value = input_flatten @ Wv + bv
  gemm_wmma_f16<<<g256, blk, 0, stream>>>(input, Wv, bv, value, MROWS, CDIM);
  // 2) off = query @ W_off + b_off
  gemm_wmma_f16<<<g256, blk, 0, stream>>>(query, W_off, b_off, off, MROWS, CDIM);
  // 3) logits = query @ W_attn + b_attn
  gemm_wmma_f16<<<g128, blk, 0, stream>>>(query, W_attn, b_attn, attn, MROWS,
                                          HN * LN * PN);
  // 4) softmax over each head's 16 logits
  softmax16<<<(MROWS * HN + 255) / 256, blk, 0, stream>>>(attn, MROWS * HN);
  // 5) deformable bilinear sampling
  deform_sample<<<MROWS, blk, 0, stream>>>(refpts, off, attn, value, samp);
  // 6) out = samp @ Wo + bo
  gemm_wmma_f16<<<g256, blk, 0, stream>>>(samp, Wo, bo, out, MROWS, CDIM);
}
